// AdaIN_VAE_61263413510362
// MI455X (gfx1250) — compile-verified, matrix operand order checked by reading, not run on hardware
//
#include <hip/hip_runtime.h>
#include <hip/hip_bf16.h>
#include <cstdint>
#include <cstddef>

typedef _Float16 v8h  __attribute__((ext_vector_type(8)));
typedef _Float16 v16h __attribute__((ext_vector_type(16)));
typedef float    v8f  __attribute__((ext_vector_type(8)));

#define AS1 __attribute__((address_space(1)))
#define AS3 __attribute__((address_space(3)))
typedef const AS1 _Float16* glb_cf16p;
typedef const AS3 _Float16* lds_cf16p;
typedef AS3 _Float16*       lds_f16p;
typedef const AS3 float*    lds_cf32p;

#define B_ 512
#define T_ 50
#define GBT 64   // batch-rows per workgroup tile in GRU kernels

// ---------------------------------------------------------------------------
// WMMA helpers: 16x16x32 f16 -> f32. A: 16(M)x32(K), B(=W rows): 16(N)x32(K).
// Per ISA layout: lane L -> row (L&15); VGPR pairs hold K = hi*8 + 0..7 and
// 16 + hi*8 + 0..7 where hi = L>>4. Two aligned 16B chunk loads each,
// emitted as global_load_b128 or ds_load_b128 depending on address space.
// ---------------------------------------------------------------------------
static __device__ __forceinline__ v16h load_frag(glb_cf16p base, int row0,
                                                 int ld, int k0, int lane) {
  int r  = row0 + (lane & 15);
  int hi = (lane >> 4) << 3;
  glb_cf16p p = base + (size_t)r * (size_t)ld + (size_t)(k0 + hi);
  union { v16h v; v8h h[2]; } u;
  u.h[0] = *(const AS1 v8h*)(p);
  u.h[1] = *(const AS1 v8h*)(p + 16);
  return u.v;
}

static __device__ __forceinline__ v16h load_frag(lds_cf16p base, int row0,
                                                 int ld, int k0, int lane) {
  int r  = row0 + (lane & 15);
  int hi = (lane >> 4) << 3;
  lds_cf16p p = base + r * ld + k0 + hi;
  union { v16h v; v8h h[2]; } u;
  u.h[0] = *(const AS3 v8h*)(p);
  u.h[1] = *(const AS3 v8h*)(p + 16);
  return u.v;
}

static __device__ __forceinline__ v8f wmma16(v16h a, v16h b, v8f c) {
  return __builtin_amdgcn_wmma_f32_16x16x32_f16(false, a, false, b,
                                                (short)0, c, false, false);
}

static __device__ __forceinline__ v8f splat8(float v) {
  v8f r;
#pragma unroll
  for (int j = 0; j < 8; ++j) r[j] = v;
  return r;
}

static __device__ __forceinline__ float sigmoidf_(float x) {
  return 1.0f / (1.0f + __expf(-x));
}

// ---------------------------------------------------------------------------
// One GRU cell step for a 16-row batch tile, all 384 gate columns, one wave.
//   gi = x @ wih^T   (K=64)      gh = h @ whh^T   (K=128)
//   r = sig(gi_r+gh_r+br), z = sig(...), n = tanh(gi_n + bihn + r*(gh_n+bhhn))
//   h' = (1-z)*n + z*h        (new h staged in regs, written after all GEMMs)
// XP is glb_cf16p (encoder: emb in global) or lds_cf16p (decoder: in-buffers).
// ---------------------------------------------------------------------------
template <typename XP>
static __device__ void gru_tile_step(XP xbase, int xRow0, int xld,
                                     lds_f16p htile,  // LDS: 16 rows, ld=128
                                     glb_cf16p wih, glb_cf16p whh,
                                     lds_cf32p bih, lds_cf32p bhh,
                                     int lane) {
  v8f rA[8], zA[8], hN[8];
  const int cl  = lane & 15;
  const int rsh = (lane >> 4) << 3;
  lds_cf16p hread = (lds_cf16p)htile;
#pragma unroll
  for (int sec = 0; sec < 2; ++sec) {           // sec 0 = r, sec 1 = z
    v8f* dst = sec ? zA : rA;
    for (int nt = 0; nt < 8; ++nt) {
      int wrow = sec * 128 + nt * 16;
      int bi = wrow + cl;
      v8f acc = splat8(bih[bi] + bhh[bi]);
      acc = wmma16(load_frag(xbase, xRow0, xld, 0, lane),
                   load_frag(wih, wrow, 64, 0, lane), acc);
      acc = wmma16(load_frag(xbase, xRow0, xld, 32, lane),
                   load_frag(wih, wrow, 64, 32, lane), acc);
#pragma unroll
      for (int kk = 0; kk < 128; kk += 32)
        acc = wmma16(load_frag(hread, 0, 128, kk, lane),
                     load_frag(whh, wrow, 128, kk, lane), acc);
#pragma unroll
      for (int j = 0; j < 8; ++j) dst[nt][j] = sigmoidf_(acc[j]);
    }
  }
  for (int nt = 0; nt < 8; ++nt) {              // n gate + combine
    int wrow = 256 + nt * 16;
    int bi = wrow + cl;
    v8f ai = splat8(bih[bi]);
    ai = wmma16(load_frag(xbase, xRow0, xld, 0, lane),
                load_frag(wih, wrow, 64, 0, lane), ai);
    ai = wmma16(load_frag(xbase, xRow0, xld, 32, lane),
                load_frag(wih, wrow, 64, 32, lane), ai);
    v8f ah = splat8(bhh[bi]);
#pragma unroll
    for (int kk = 0; kk < 128; kk += 32)
      ah = wmma16(load_frag(hread, 0, 128, kk, lane),
                  load_frag(whh, wrow, 128, kk, lane), ah);
#pragma unroll
    for (int j = 0; j < 8; ++j) {
      float hold = (float)hread[(j + rsh) * 128 + nt * 16 + cl];
      float nv = tanhf(ai[j] + rA[nt][j] * ah[j]);
      float zz = zA[nt][j];
      hN[nt][j] = (1.0f - zz) * nv + zz * hold;
    }
  }
#pragma unroll
  for (int nt = 0; nt < 8; ++nt)
#pragma unroll
    for (int j = 0; j < 8; ++j)
      htile[(j + rsh) * 128 + nt * 16 + cl] = (_Float16)hN[nt][j];
}

// ---------------------------------------------------------------------------
// Encoder GRU scan: grid.x * GBT batch rows, 4 waves (128 thr), h in LDS.
// emb layout [Nb, T, 64] f16 (row stride T*64). Writes final h (f16) into
// hcat[Nb,256] at colOff (0 = forward hT, 128 = backward hB slot).
// ---------------------------------------------------------------------------
__global__ void __launch_bounds__(128) gru_scan_kernel(
    const _Float16* emb, int t0, int t1,
    const _Float16* wih, const _Float16* whh,
    const float* bihG, const float* bhhG,
    _Float16* hcat, int colOff) {
  __shared__ __align__(16) _Float16 hbuf[GBT][128];
  __shared__ float bihL[384], bhhL[384];
  int tid = threadIdx.x;
  for (int i = tid; i < 384; i += 128) { bihL[i] = bihG[i]; bhhL[i] = bhhG[i]; }
  for (int i = tid; i < GBT * 128; i += 128) ((_Float16*)hbuf)[i] = (_Float16)0.0f;
  __syncthreads();
  int wave = tid >> 5, lane = tid & 31;
  int mBase = wave * 16;
  int gRow0 = blockIdx.x * GBT;
  for (int t = t0; t < t1; ++t) {
    glb_cf16p xb = (glb_cf16p)(emb + (size_t)t * 64);
    gru_tile_step(xb, gRow0 + mBase, T_ * 64, (lds_f16p)&hbuf[mBase][0],
                  (glb_cf16p)wih, (glb_cf16p)whh,
                  (lds_cf32p)bihL, (lds_cf32p)bhhL, lane);
  }
  __syncthreads();
  for (int i = tid; i < GBT * 128; i += 128) {
    int r = i >> 7, c = i & 127;
    hcat[(size_t)(gRow0 + r) * 256 + colOff + c] = hbuf[r][c];
  }
}

// ---------------------------------------------------------------------------
// Fused double-GRU decoder: 8 waves = 2 cells x 4 M-tiles, 64 batch rows/WG.
// Per step: WMMA GRU for both cells, then VALU stages for o (H->2),
// e (2->32) and hi (H->32) feeding next step's in1/in2 (LDS, f16).
// ---------------------------------------------------------------------------
__global__ void __launch_bounds__(256) gru_decode_kernel(
    const _Float16* h0, const _Float16* in1g, const _Float16* in2g,
    const _Float16* wih1, const _Float16* whh1, const float* bih1g, const float* bhh1g,
    const _Float16* wih2, const _Float16* whh2, const float* bih2g, const float* bhh2g,
    const float* dec1w, const float* dec1b, const float* dec2w, const float* dec2b,
    const float* ed1w, const float* ed1b, const float* ed2w, const float* ed2b,
    const float* h2i1w, const float* h2i1b, const float* h2i2w, const float* h2i2b,
    float* outs) {
  __shared__ __align__(16) _Float16 hls[2][GBT][128];
  __shared__ __align__(16) _Float16 inb[2][GBT][64];
  __shared__ float bihL[2][384], bhhL[2][384];
  __shared__ float oL[2][GBT][2];
  int tid = threadIdx.x;
  int gRow0 = blockIdx.x * GBT;
  for (int i = tid; i < 384; i += 256) {
    bihL[0][i] = bih1g[i]; bhhL[0][i] = bhh1g[i];
    bihL[1][i] = bih2g[i]; bhhL[1][i] = bhh2g[i];
  }
  for (int i = tid; i < GBT * 128; i += 256) {
    int r = i >> 7, c = i & 127;
    _Float16 v = h0[(size_t)(gRow0 + r) * 128 + c];
    hls[0][r][c] = v; hls[1][r][c] = v;
  }
  for (int i = tid; i < GBT * 64; i += 256) {
    int r = i >> 6, c = i & 63;
    inb[0][r][c] = in1g[(size_t)(gRow0 + r) * 64 + c];
    inb[1][r][c] = in2g[(size_t)(gRow0 + r) * 64 + c];
  }
  __syncthreads();
  int wave = tid >> 5, lane = tid & 31;
  int cell = wave >> 2;
  int mBase = (wave & 3) * 16;
  const _Float16* wih = cell ? wih2 : wih1;
  const _Float16* whh = cell ? whh2 : whh1;
  for (int t = 0; t < T_; ++t) {
    gru_tile_step((lds_cf16p)&inb[cell][mBase][0], 0, 64,
                  (lds_f16p)&hls[cell][mBase][0],
                  (glb_cf16p)wih, (glb_cf16p)whh,
                  (lds_cf32p)bihL[cell], (lds_cf32p)bhhL[cell], lane);
    __syncthreads();
    {  // o stage: 256 threads -> (cell, row, col) of the [64,2] projections
      int r = tid >> 2, q = tid & 3;
      int cc = q >> 1, co = q & 1;
      const float* dw = cc ? dec2w : dec1w;
      float o = (cc ? dec2b : dec1b)[co];
      for (int k = 0; k < 128; ++k) o += (float)hls[cc][r][k] * dw[co * 128 + k];
      oL[cc][r][co] = o;
      outs[(size_t)(gRow0 + r) * (T_ * 4) + t * 4 + cc * 2 + co] = o;
    }
    __syncthreads();
    for (int it = 0; it < 16; ++it) {  // e + hi stages
      int idx = tid + it * 256;
      int cc = idx >> 11;
      int rem = idx & 2047;
      int r = rem >> 5, c = rem & 31;
      const float* edw = cc ? ed2w : ed1w;
      float e = (cc ? ed2b : ed1b)[c] + oL[cc][r][0] * edw[c * 2]
                                      + oL[cc][r][1] * edw[c * 2 + 1];
      inb[cc][r][c] = (_Float16)e;  // e1->in1, e2->in2 (low 32 cols)
      const float* hw = cc ? h2i2w : h2i1w;
      float acc = (cc ? h2i2b : h2i1b)[c];
      for (int k = 0; k < 128; ++k) acc += (float)hls[cc][r][k] * hw[c * 128 + k];
      inb[cc ^ 1][r][32 + c] = (_Float16)acc;  // hi2->in1, hi1->in2 (high 32)
    }
    __syncthreads();
  }
}

// ---------------------------------------------------------------------------
// Generic WMMA FC: out = act(A_f16[M,K] @ W_f16[N,K]^T + bias). One 16x16 tile
// per wave, 8 waves per WG; grid.x = (M/16)*(N/16)/8 (all uses divide evenly).
// ---------------------------------------------------------------------------
__global__ void __launch_bounds__(256) fc_wmma_kernel(
    const _Float16* A, int lda, const _Float16* W, const float* bias,
    float* o32, int ldo32, _Float16* o16, int ldo16, int colOff,
    int nTiles, int K, int act) {
  int tile = blockIdx.x * 8 + (threadIdx.x >> 5);
  int mt = tile / nTiles, nt = tile % nTiles;
  int lane = threadIdx.x & 31;
  int cl = lane & 15, rsh = (lane >> 4) << 3;
  int col = nt * 16 + cl;
  v8f acc = splat8(bias ? bias[col] : 0.0f);
  glb_cf16p Ag = (glb_cf16p)A;
  glb_cf16p Wg = (glb_cf16p)W;
  for (int k = 0; k < K; k += 32)
    acc = wmma16(load_frag(Ag, mt * 16, lda, k, lane),
                 load_frag(Wg, nt * 16, K, k, lane), acc);
#pragma unroll
  for (int j = 0; j < 8; ++j) {
    float v = acc[j];
    if (act) v = fmaxf(v, 0.0f);
    int row = mt * 16 + j + rsh;
    if (o32) o32[(size_t)row * ldo32 + col] = v;
    if (o16) o16[(size_t)row * ldo16 + colOff + col] = (_Float16)v;
  }
}

// --------------------------- small utility kernels -------------------------
__global__ void cvt_f32_f16_kernel(const float* s, _Float16* d, int n) {
  int i = blockIdx.x * 256 + threadIdx.x;
  if (i < n) d[i] = (_Float16)s[i];
}

__global__ void embed_kernel(const float* x, const float* W, const float* b,
                             _Float16* dst, long long dstOffHalfs, int nbT) {
  int bt = blockIdx.x;
  int e = threadIdx.x;
  if (bt >= nbT) return;
  const float* xr = x + (size_t)bt * 4;
  float v = b[e] + xr[0] * W[e * 4 + 0] + xr[1] * W[e * 4 + 1]
                 + xr[2] * W[e * 4 + 2] + xr[3] * W[e * 4 + 3];
  dst[(size_t)dstOffHalfs + (size_t)bt * 64 + e] = (_Float16)v;
}

__global__ void conv3x3s2_kernel(const float* in32, const _Float16* in16,
                                 int NB, int Ci, int Hin, int Co, int Hout,
                                 const float* W, const float* bias,
                                 _Float16* out, int nOff) {
  size_t idx = (size_t)blockIdx.x * 256 + threadIdx.x;
  size_t total = (size_t)NB * Co * Hout * Hout;
  if (idx >= total) return;
  int ox = (int)(idx % Hout); size_t t = idx / Hout;
  int oy = (int)(t % Hout); t /= Hout;
  int co = (int)(t % Co); int n = (int)(t / Co);
  float acc = bias[co];
  int iy0 = oy * 2 - 1, ix0 = ox * 2 - 1;
  for (int ci = 0; ci < Ci; ++ci) {
    const float* wp = W + ((size_t)co * Ci + ci) * 9;
    size_t ib = ((size_t)n * Ci + ci) * Hin * Hin;
    for (int ky = 0; ky < 3; ++ky) {
      int iy = iy0 + ky; if (iy < 0 || iy >= Hin) continue;
      for (int kx = 0; kx < 3; ++kx) {
        int ix = ix0 + kx; if (ix < 0 || ix >= Hin) continue;
        float v = in32 ? in32[ib + (size_t)iy * Hin + ix]
                       : (float)in16[ib + (size_t)iy * Hin + ix];
        acc += v * wp[ky * 3 + kx];
      }
    }
  }
  out[((size_t)(n + nOff) * Co + co) * Hout * Hout + (size_t)oy * Hout + ox] =
      (_Float16)fmaxf(acc, 0.0f);
}

__global__ void expstd_kernel(const float* lv, float* sd, int n) {
  int i = blockIdx.x * 256 + threadIdx.x;
  if (i < n) sd[i] = __expf(0.5f * lv[i]);
}

// adain: one 64-thread block per output row (4 streams x 512 rows).
// streams: 0=z_c_s  1=z_c_t  2=z_f_s  3=z_f_t   (decode pack order)
__global__ void adain_kernel(const float* head, const float* eps_s,
                             const float* eps_t, const float* wptr,
                             const float* condm, const float* condstd,
                             float* z32, _Float16* z16) {
  int row = blockIdx.x; int s = row >> 9; int b = row & (B_ - 1);
  int c = threadIdx.x;
  float mu_s = head[(size_t)b * 128 + c],        lv_s = head[(size_t)b * 128 + 64 + c];
  float mu_t = head[(size_t)(B_ + b) * 128 + c], lv_t = head[(size_t)(B_ + b) * 128 + 64 + c];
  float zs = mu_s + __expf(0.5f * lv_s) * eps_s[b * 64 + c];
  float zt = mu_t + __expf(0.5f * lv_t) * eps_t[b * 64 + c];
  float w = wptr[0];
  float x = (s == 0) ? zs : (s == 1) ? zt : (w * zs + (1.0f - w) * zt);
  __shared__ float red[64];
  red[c] = x; __syncthreads();
  for (int st = 32; st > 0; st >>= 1) { if (c < st) red[c] += red[c + st]; __syncthreads(); }
  float mean = red[0] * (1.0f / 64.0f);
  __syncthreads();
  float d = x - mean;
  red[c] = d * d; __syncthreads();
  for (int st = 32; st > 0; st >>= 1) { if (c < st) red[c] += red[c + st]; __syncthreads(); }
  float sd = sqrtf(red[0] * (1.0f / 63.0f));  // unbiased (ddof=1)
  int cr = ((s == 0) || (s == 2)) ? b : (B_ + b);
  float out = condstd[(size_t)cr * 64 + c] * d / sd + condm[(size_t)cr * 64 + c];
  z32[(size_t)row * 64 + c] = out;
  z16[(size_t)row * 64 + c] = (_Float16)out;
}

__global__ void dec_estart_kernel(const float* xs, const float* xt,
                                  const float* ed1w, const float* ed1b,
                                  const float* ed2w, const float* ed2b,
                                  _Float16* in1, _Float16* in2) {
  int row = blockIdx.x; int s = row >> 9; int b = row & (B_ - 1);
  int c = threadIdx.x;  // 32
  const float* x = (s & 1) ? xt : xs;
  float a0 = x[(size_t)b * T_ * 4 + 0], a1 = x[(size_t)b * T_ * 4 + 1];
  float a2 = x[(size_t)b * T_ * 4 + 2], a3 = x[(size_t)b * T_ * 4 + 3];
  in1[(size_t)row * 64 + c] = (_Float16)(ed1b[c] + a0 * ed1w[c * 2] + a1 * ed1w[c * 2 + 1]);
  in2[(size_t)row * 64 + c] = (_Float16)(ed2b[c] + a2 * ed2w[c * 2] + a3 * ed2w[c * 2 + 1]);
}

__global__ void gather_kernel(const float* douts, const float* headA,
                              const float* headB, const float* zall, float* out) {
  int b = blockIdx.x;
  for (int col = threadIdx.x; col < 1168; col += 256) {
    float v;
    if (col < 400) {
      int s = col / 200;
      int k = col - s * 200;
      v = douts[((size_t)(s * B_ + b)) * 200 + k];
    } else {
      int p = (col - 400) >> 6;
      int k = (col - 400) & 63;
      switch (p) {
        case 0:  v = headA[(size_t)b * 128 + k]; break;                 // mu_s
        case 1:  v = headA[(size_t)b * 128 + 64 + k]; break;            // lv_s
        case 2:  v = headA[(size_t)(B_ + b) * 128 + k]; break;          // mu_t
        case 3:  v = headA[(size_t)(B_ + b) * 128 + 64 + k]; break;     // lv_t
        case 4:  v = zall[(size_t)(2 * B_ + b) * 64 + k]; break;        // z_f_s
        case 5:  v = headB[(size_t)(2 * B_ + b) * 128 + k]; break;      // mu_f_s
        case 6:  v = zall[(size_t)(3 * B_ + b) * 64 + k]; break;        // z_f_t
        case 7:  v = headB[(size_t)(3 * B_ + b) * 128 + k]; break;      // mu_f_t
        case 8:  v = zall[(size_t)(0 * B_ + b) * 64 + k]; break;        // z_c_s
        case 9:  v = headB[(size_t)(0 * B_ + b) * 128 + k]; break;      // mu_s_new
        case 10: v = zall[(size_t)(1 * B_ + b) * 64 + k]; break;        // z_c_t
        default: v = headB[(size_t)(1 * B_ + b) * 128 + k]; break;      // mu_t_new
      }
    }
    out[(size_t)b * 1168 + col] = v;
  }
}

// ===========================================================================
extern "C" void kernel_launch(void* const* d_in, const int* in_sizes, int n_in,
                              void* d_out, int out_size, void* d_ws, size_t ws_size,
                              hipStream_t stream) {
  (void)in_sizes; (void)n_in; (void)out_size; (void)ws_size;
  // ---- param indices (dict insertion order of make_params, then inputs) ----
  const float* emb_w  = (const float*)d_in[0];
  const float* emb_b  = (const float*)d_in[1];
  const float* z2h_w  = (const float*)d_in[2];
  const float* z2h_b  = (const float*)d_in[3];
  const float* enc_w  = (const float*)d_in[4];
  const float* enc_bb = (const float*)d_in[5];
  const float* dec1w  = (const float*)d_in[6];
  const float* dec1b  = (const float*)d_in[7];
  const float* dec2w  = (const float*)d_in[8];
  const float* dec2b  = (const float*)d_in[9];
  const float* ed1w   = (const float*)d_in[10];
  const float* ed1b   = (const float*)d_in[11];
  const float* ed2w   = (const float*)d_in[12];
  const float* ed2b   = (const float*)d_in[13];
  const float* h2i1w  = (const float*)d_in[14];
  const float* h2i1b  = (const float*)d_in[15];
  const float* h2i2w  = (const float*)d_in[16];
  const float* h2i2b  = (const float*)d_in[17];
  const float* ef_wih = (const float*)d_in[18];
  const float* ef_whh = (const float*)d_in[19];
  const float* ef_bih = (const float*)d_in[20];
  const float* ef_bhh = (const float*)d_in[21];
  const float* eb_wih = (const float*)d_in[22];
  const float* eb_whh = (const float*)d_in[23];
  const float* eb_bih = (const float*)d_in[24];
  const float* eb_bhh = (const float*)d_in[25];
  const float* d1_wih = (const float*)d_in[26];
  const float* d1_whh = (const float*)d_in[27];
  const float* d1_bih = (const float*)d_in[28];
  const float* d1_bhh = (const float*)d_in[29];
  const float* d2_wih = (const float*)d_in[30];
  const float* d2_whh = (const float*)d_in[31];
  const float* d2_bih = (const float*)d_in[32];
  const float* d2_bhh = (const float*)d_in[33];
  const float* c0w = (const float*)d_in[34]; const float* c0b = (const float*)d_in[35];
  const float* c1w = (const float*)d_in[36]; const float* c1b = (const float*)d_in[37];
  const float* c2w = (const float*)d_in[38]; const float* c2b = (const float*)d_in[39];
  const float* c3w = (const float*)d_in[40]; const float* c3b = (const float*)d_in[41];
  const float* cm1w = (const float*)d_in[42]; const float* cm1b = (const float*)d_in[43];
  const float* cm2w = (const float*)d_in[44]; const float* cm2b = (const float*)d_in[45];
  const float* cl1w = (const float*)d_in[46]; const float* cl1b = (const float*)d_in[47];
  const float* cl2w = (const float*)d_in[48]; const float* cl2b = (const float*)d_in[49];
  const float* x_s  = (const float*)d_in[50];
  const float* x_t  = (const float*)d_in[51];
  const float* c_s  = (const float*)d_in[52];
  const float* c_t  = (const float*)d_in[53];
  const float* eps_s = (const float*)d_in[54];
  const float* eps_t = (const float*)d_in[55];
  const float* wgt   = (const float*)d_in[56];
  float* out = (float*)d_out;

  // ---- workspace carve-up ----
  char* ws = (char*)d_ws;
  size_t off = 0;
  auto alloc = [&](size_t bytes) -> char* {
    char* p = ws + off;
    off = (off + bytes + 255) & ~(size_t)255;
    return p;
  };
  _Float16* ef_wih16 = (_Float16*)alloc(384 * 64 * 2);
  _Float16* ef_whh16 = (_Float16*)alloc(384 * 128 * 2);
  _Float16* eb_wih16 = (_Float16*)alloc(384 * 64 * 2);
  _Float16* eb_whh16 = (_Float16*)alloc(384 * 128 * 2);
  _Float16* d1_wih16 = (_Float16*)alloc(384 * 64 * 2);
  _Float16* d1_whh16 = (_Float16*)alloc(384 * 128 * 2);
  _Float16* d2_wih16 = (_Float16*)alloc(384 * 64 * 2);
  _Float16* d2_whh16 = (_Float16*)alloc(384 * 128 * 2);
  _Float16* enc_w16  = (_Float16*)alloc(128 * 256 * 2);
  _Float16* z2h_w16  = (_Float16*)alloc(128 * 64 * 2);
  _Float16* h2i1w16  = (_Float16*)alloc(32 * 128 * 2);
  _Float16* h2i2w16  = (_Float16*)alloc(32 * 128 * 2);
  _Float16* cm1w16   = (_Float16*)alloc((size_t)256 * 4096 * 2);
  _Float16* cm2w16   = (_Float16*)alloc(64 * 256 * 2);
  _Float16* cl1w16   = (_Float16*)alloc((size_t)256 * 4096 * 2);
  _Float16* cl2w16   = (_Float16*)alloc(64 * 256 * 2);
  _Float16* emb_a  = (_Float16*)alloc((size_t)1024 * T_ * 64 * 2);
  _Float16* emb_b2 = (_Float16*)alloc((size_t)2048 * T_ * 64 * 2);
  _Float16* hcat_a = (_Float16*)alloc((size_t)1024 * 256 * 2);
  _Float16* hcat_b = (_Float16*)alloc((size_t)2048 * 256 * 2);
  float*    head_a = (float*)alloc((size_t)1024 * 128 * 4);
  float*    head_b = (float*)alloc((size_t)2048 * 128 * 4);
  _Float16* a0 = (_Float16*)alloc((size_t)1024 * 16 * 64 * 64 * 2);
  _Float16* a1 = (_Float16*)alloc((size_t)1024 * 32 * 32 * 32 * 2);
  _Float16* a2 = (_Float16*)alloc((size_t)1024 * 64 * 16 * 16 * 2);
  _Float16* a3 = (_Float16*)alloc((size_t)1024 * 4096 * 2);
  _Float16* fcm = (_Float16*)alloc((size_t)1024 * 256 * 2);
  _Float16* fcl = (_Float16*)alloc((size_t)1024 * 256 * 2);
  float* condm  = (float*)alloc((size_t)1024 * 64 * 4);
  float* condlv = (float*)alloc((size_t)1024 * 64 * 4);
  float* condsd = (float*)alloc((size_t)1024 * 64 * 4);
  float*    zall32 = (float*)alloc((size_t)2048 * 64 * 4);
  _Float16* zall16 = (_Float16*)alloc((size_t)2048 * 64 * 2);
  _Float16* h0d = (_Float16*)alloc((size_t)2048 * 128 * 2);
  _Float16* in1 = (_Float16*)alloc((size_t)2048 * 64 * 2);
  _Float16* in2 = (_Float16*)alloc((size_t)2048 * 64 * 2);
  float* douts = (float*)alloc((size_t)2048 * T_ * 4 * 4);

  auto cvt = [&](const float* s, _Float16* d, int n) {
    cvt_f32_f16_kernel<<<dim3((n + 255) / 256), dim3(256), 0, stream>>>(s, d, n);
  };
  // generic fc launcher
  auto fc = [&](const _Float16* A, int lda, const _Float16* W, const float* bias,
                float* o32, int ldo32, _Float16* o16, int ldo16, int colOff,
                int M, int N, int K, int act) {
    int tiles = (M / 16) * (N / 16);
    fc_wmma_kernel<<<dim3(tiles / 8), dim3(256), 0, stream>>>(
        A, lda, W, bias, o32, ldo32, o16, ldo16, colOff, N / 16, K, act);
  };

  // ---- phase 0: weight conversions ----
  cvt(ef_wih, ef_wih16, 384 * 64);  cvt(ef_whh, ef_whh16, 384 * 128);
  cvt(eb_wih, eb_wih16, 384 * 64);  cvt(eb_whh, eb_whh16, 384 * 128);
  cvt(d1_wih, d1_wih16, 384 * 64);  cvt(d1_whh, d1_whh16, 384 * 128);
  cvt(d2_wih, d2_wih16, 384 * 64);  cvt(d2_whh, d2_whh16, 384 * 128);
  cvt(enc_w, enc_w16, 128 * 256);   cvt(z2h_w, z2h_w16, 128 * 64);
  cvt(h2i1w, h2i1w16, 32 * 128);    cvt(h2i2w, h2i2w16, 32 * 128);
  cvt(cm1w, cm1w16, 256 * 4096);    cvt(cm2w, cm2w16, 64 * 256);
  cvt(cl1w, cl1w16, 256 * 4096);    cvt(cl2w, cl2w16, 64 * 256);

  // ---- phase 1: embeddings (x_s|x_t packed) + conv stack (c_s|c_t packed) --
  embed_kernel<<<dim3(B_ * T_), dim3(64), 0, stream>>>(x_s, emb_w, emb_b, emb_a, 0, B_ * T_);
  embed_kernel<<<dim3(B_ * T_), dim3(64), 0, stream>>>(x_t, emb_w, emb_b, emb_a,
                                                       (long long)B_ * T_ * 64, B_ * T_);
  {
    size_t tot0 = (size_t)B_ * 16 * 64 * 64;
    conv3x3s2_kernel<<<dim3((tot0 + 255) / 256), dim3(256), 0, stream>>>(
        c_s, nullptr, B_, 1, 128, 16, 64, c0w, c0b, a0, 0);
    conv3x3s2_kernel<<<dim3((tot0 + 255) / 256), dim3(256), 0, stream>>>(
        c_t, nullptr, B_, 1, 128, 16, 64, c0w, c0b, a0, B_);
    size_t tot1 = (size_t)1024 * 32 * 32 * 32;
    conv3x3s2_kernel<<<dim3((tot1 + 255) / 256), dim3(256), 0, stream>>>(
        nullptr, a0, 1024, 16, 64, 32, 32, c1w, c1b, a1, 0);
    size_t tot2 = (size_t)1024 * 64 * 16 * 16;
    conv3x3s2_kernel<<<dim3((tot2 + 255) / 256), dim3(256), 0, stream>>>(
        nullptr, a1, 1024, 32, 32, 64, 16, c2w, c2b, a2, 0);
    size_t tot3 = (size_t)1024 * 64 * 8 * 8;
    conv3x3s2_kernel<<<dim3((tot3 + 255) / 256), dim3(256), 0, stream>>>(
        nullptr, a2, 1024, 64, 16, 64, 8, c3w, c3b, a3, 0);
  }

  // ---- phase 2: first encodes + cond heads ----
  gru_scan_kernel<<<dim3(1024 / GBT), dim3(128), 0, stream>>>(
      emb_a, 0, T_, ef_wih16, ef_whh16, ef_bih, ef_bhh, hcat_a, 0);
  gru_scan_kernel<<<dim3(1024 / GBT), dim3(128), 0, stream>>>(
      emb_a, T_ - 1, T_, eb_wih16, eb_whh16, eb_bih, eb_bhh, hcat_a, 128);
  fc(hcat_a, 256, enc_w16, enc_bb, head_a, 128, nullptr, 0, 0, 1024, 128, 256, 0);
  fc(a3, 4096, cm1w16, cm1b, nullptr, 0, fcm, 256, 0, 1024, 256, 4096, 1);
  fc(fcm, 256, cm2w16, cm2b, condm, 64, nullptr, 0, 0, 1024, 64, 256, 0);
  fc(a3, 4096, cl1w16, cl1b, nullptr, 0, fcl, 256, 0, 1024, 256, 4096, 1);
  fc(fcl, 256, cl2w16, cl2b, condlv, 64, nullptr, 0, 0, 1024, 64, 256, 0);
  expstd_kernel<<<dim3((1024 * 64 + 255) / 256), dim3(256), 0, stream>>>(
      condlv, condsd, 1024 * 64);

  // ---- phase 3: z sampling + adain (4 streams packed) ----
  adain_kernel<<<dim3(2048), dim3(64), 0, stream>>>(
      head_a, eps_s, eps_t, wgt, condm, condsd, zall32, zall16);

  // ---- phase 4: decoder init + fused decode ----
  fc(zall16, 64, z2h_w16, z2h_b, nullptr, 0, h0d, 128, 0, 2048, 128, 64, 0);
  fc(h0d, 128, h2i2w16, h2i2b, nullptr, 0, in1, 64, 32, 2048, 32, 128, 0);
  fc(h0d, 128, h2i1w16, h2i1b, nullptr, 0, in2, 64, 32, 2048, 32, 128, 0);
  dec_estart_kernel<<<dim3(2048), dim3(32), 0, stream>>>(
      x_s, x_t, ed1w, ed1b, ed2w, ed2b, in1, in2);
  gru_decode_kernel<<<dim3(2048 / GBT), dim3(256), 0, stream>>>(
      h0d, in1, in2,
      d1_wih16, d1_whh16, d1_bih, d1_bhh,
      d2_wih16, d2_whh16, d2_bih, d2_bhh,
      dec1w, dec1b, dec2w, dec2b, ed1w, ed1b, ed2w, ed2b,
      h2i1w, h2i1b, h2i2w, h2i2b, douts);

  // ---- phase 5: re-encode all 4 decoded streams ----
  embed_kernel<<<dim3(2048 * T_), dim3(64), 0, stream>>>(
      douts, emb_w, emb_b, emb_b2, 0, 2048 * T_);
  gru_scan_kernel<<<dim3(2048 / GBT), dim3(128), 0, stream>>>(
      emb_b2, 0, T_, ef_wih16, ef_whh16, ef_bih, ef_bhh, hcat_b, 0);
  gru_scan_kernel<<<dim3(2048 / GBT), dim3(128), 0, stream>>>(
      emb_b2, T_ - 1, T_, eb_wih16, eb_whh16, eb_bih, eb_bhh, hcat_b, 128);
  fc(hcat_b, 256, enc_w16, enc_bb, head_b, 128, nullptr, 0, 0, 2048, 128, 256, 0);

  // ---- phase 6: assemble output [512, 1168] ----
  gather_kernel<<<dim3(B_), dim3(256), 0, stream>>>(douts, head_a, head_b, zall32, out);
}
